// EncoderLayer_78460462564046
// MI455X (gfx1250) — matrix operand order repaired and checked by reading, not compiled, not run on hardware
//
#include <hip/hip_runtime.h>
#include <hip/hip_bf16.h>
#include <math.h>

// ---------------------------------------------------------------------------
// Transformer encoder layer for MI455X (gfx1250, wave32, WMMA).
// All matrix math via v_wmma_f32_16x16x32_bf16. All WMMA operands are kept
// in bf16 global buffers so tile staging is a pure 16B copy -> eligible for
// the CDNA5 async load-to-LDS path (ASYNCcnt) when the builtin exists.
// ---------------------------------------------------------------------------

typedef __attribute__((ext_vector_type(16))) __bf16        v16bf;
typedef __attribute__((ext_vector_type(8)))  float         v8f;
typedef __attribute__((ext_vector_type(8)))  unsigned int  v8u;
typedef __attribute__((ext_vector_type(4)))  int           v4i;

typedef __attribute__((address_space(1))) v4i* gv4i_p;   // global <4 x i32>*
typedef __attribute__((address_space(3))) v4i* lv4i_p;   // LDS    <4 x i32>*

#if __has_builtin(__builtin_amdgcn_global_load_async_to_lds_b128) && \
    __has_builtin(__builtin_amdgcn_s_wait_asynccnt)
#define HAVE_ASYNC 1
#else
#define HAVE_ASYNC 0
#endif

__device__ __forceinline__ unsigned short f2bf(float f) {
    unsigned u = __builtin_bit_cast(unsigned, f);
    unsigned r = u + 0x7FFFu + ((u >> 16) & 1u);   // round-to-nearest-even
    return (unsigned short)(r >> 16);
}

__device__ __forceinline__ void st4bf(unsigned short* dst, float4 v) {
    unsigned lo = (unsigned)f2bf(v.x) | ((unsigned)f2bf(v.y) << 16);
    unsigned hi = (unsigned)f2bf(v.z) | ((unsigned)f2bf(v.w) << 16);
    *reinterpret_cast<uint2*>(dst) = make_uint2(lo, hi);
}

__device__ __forceinline__ v8f zero8() {
    v8f z; for (int i = 0; i < 8; ++i) z[i] = 0.0f; return z;
}

// 16-byte global -> LDS copy. Uses gfx1250 async-to-LDS when available.
__device__ __forceinline__ void cp16(unsigned short* ldst,
                                     const unsigned short* gsrc) {
#if HAVE_ASYNC
    __builtin_amdgcn_global_load_async_to_lds_b128(
        (gv4i_p)(unsigned long long)gsrc,
        (lv4i_p)(unsigned int)(unsigned long long)ldst, 0, 0);
#else
    *reinterpret_cast<uint4*>(ldst) = *reinterpret_cast<const uint4*>(gsrc);
#endif
}

__device__ __forceinline__ void async_wait_all() {
#if HAVE_ASYNC
    __builtin_amdgcn_s_wait_asynccnt(0);
#endif
}

// A-matrix 16x32 bf16 fragment: m = lane&15,
// lanes 0-15 hold K=0..15, lanes 16-31 hold K=16..31.
__device__ __forceinline__ v16bf ld_fragA(const unsigned short* base, int row0,
                                          int ld, int koff) {
    int lane = threadIdx.x & 31;
    const unsigned short* p =
        base + (row0 + (lane & 15)) * ld + koff + ((lane >> 4) << 4);
    uint4 a = *reinterpret_cast<const uint4*>(p);
    uint4 b = *reinterpret_cast<const uint4*>(p + 8);
    v8u u;
    u[0]=a.x; u[1]=a.y; u[2]=a.z; u[3]=a.w;
    u[4]=b.x; u[5]=b.y; u[6]=b.z; u[7]=b.w;
    return __builtin_bit_cast(v16bf, u);
}

// B-matrix 32x16 bf16 fragment: n = lane&15, lanes 0-15 hold K=0..15,
// lanes 16-31 hold K=16..31. base is row-per-N (i.e. B^T) in LDS.
__device__ __forceinline__ v16bf ld_fragB(const unsigned short* base, int row0,
                                          int ld, int koff) {
    int lane = threadIdx.x & 31;
    const unsigned short* p =
        base + (row0 + (lane & 15)) * ld + koff + ((lane >> 4) << 4);
    uint4 a = *reinterpret_cast<const uint4*>(p);
    uint4 b = *reinterpret_cast<const uint4*>(p + 8);
    v8u u;
    u[0]=a.x; u[1]=a.y; u[2]=a.z; u[3]=a.w;
    u[4]=b.x; u[5]=b.y; u[6]=b.z; u[7]=b.w;
    return __builtin_bit_cast(v16bf, u);
}

__device__ __forceinline__ v8f wmma_bf16(v16bf a, v16bf b, v8f c) {
    return __builtin_amdgcn_wmma_f32_16x16x32_bf16(
        false, a, false, b, (short)0, c, false, false);
}

// ---------------------------------------------------------------------------
// fp32 -> bf16 bulk convert (4 elements/thread).
// ---------------------------------------------------------------------------
__global__ __launch_bounds__(256)
void cvt_bf16_kernel(const float* __restrict__ src,
                     unsigned short* __restrict__ dst) {
    size_t i = ((size_t)blockIdx.x * 256 + threadIdx.x) * 4;
    float4 v = *reinterpret_cast<const float4*>(src + i);
    st4bf(dst + i, v);
}

// ---------------------------------------------------------------------------
// W[K,N] f32 -> Wt[N,K] bf16 (32x32 LDS tile transpose).
// ---------------------------------------------------------------------------
__global__ __launch_bounds__(256)
void transpose_w_kernel(const float* __restrict__ W,
                        unsigned short* __restrict__ Wt, int K, int N) {
    __shared__ float t[32][33];
    const int k0 = blockIdx.y * 32, n0 = blockIdx.x * 32;
    const int tx = threadIdx.x & 31, ty = threadIdx.x >> 5;
    for (int j = 0; j < 4; ++j)
        t[ty + 8 * j][tx] = W[(size_t)(k0 + ty + 8 * j) * N + n0 + tx];
    __syncthreads();
    for (int j = 0; j < 4; ++j)
        Wt[(size_t)(n0 + ty + 8 * j) * K + k0 + tx] = f2bf(t[tx][ty + 8 * j]);
}

// ---------------------------------------------------------------------------
// GEMM: C = A[M,K](bf16) * Bt[N,K]^T(bf16) + bias.
// Block 256 threads (8 waves), tile 256x128, per-wave 64x64 (4x4 frags),
// BK=32. Outputs (any may be null): Cf (f32), Cb (bf16, row-major),
// Ct (bf16, transposed: Ct[n*M + m]).
// ---------------------------------------------------------------------------
#define BMG 256
#define BNG 128
#define BKD 32
#define LDA 40
#define LDB 40

__global__ __launch_bounds__(256)
void gemm_bf_kernel(const unsigned short* __restrict__ A,
                    const unsigned short* __restrict__ Bt,
                    const float* __restrict__ bias,
                    float* __restrict__ Cf,
                    unsigned short* __restrict__ Cb,
                    unsigned short* __restrict__ Ct,
                    int M, int N, int K, int relu) {
    __shared__ unsigned short a_lds[BMG * LDA];
    __shared__ unsigned short bt_lds[BNG * LDB];

    const int tid  = threadIdx.x;
    const int lane = tid & 31;
    const int wave = tid >> 5;
    const int wm   = wave >> 1;   // 4 strips of 64 rows
    const int wn   = wave & 1;    // 2 strips of 64 cols
    const int m0   = blockIdx.y * BMG;
    const int n0   = blockIdx.x * BNG;

    v8f acc[4][4];
    for (int mi = 0; mi < 4; ++mi)
        for (int nj = 0; nj < 4; ++nj) acc[mi][nj] = zero8();

    for (int k0 = 0; k0 < K; k0 += BKD) {
        // Stage A 256x32 and Bt 128x32 as pure 16B copies.
        for (int i = tid; i < 1024; i += 256) {
            int r = i >> 2, c8 = (i & 3) << 3;
            cp16(&a_lds[r * LDA + c8], A + (size_t)(m0 + r) * K + k0 + c8);
        }
        for (int i = tid; i < 512; i += 256) {
            int r = i >> 2, c8 = (i & 3) << 3;
            cp16(&bt_lds[r * LDB + c8], Bt + (size_t)(n0 + r) * K + k0 + c8);
        }
        async_wait_all();
        __syncthreads();

        v16bf af[4], bf[4];
        for (int mi = 0; mi < 4; ++mi)
            af[mi] = ld_fragA(a_lds, wm * 64 + mi * 16, LDA, 0);
        for (int nj = 0; nj < 4; ++nj)
            bf[nj] = ld_fragB(bt_lds, wn * 64 + nj * 16, LDB, 0);
        for (int mi = 0; mi < 4; ++mi)
            for (int nj = 0; nj < 4; ++nj)
                acc[mi][nj] = wmma_bf16(af[mi], bf[nj], acc[mi][nj]);
        __syncthreads();
    }

    // Epilogue. C/D layout: N = lane&15, M = r + (lane>=16 ? 8 : 0).
    const int hi8 = (lane >> 4) << 3;
    const int cn  = lane & 15;
    for (int nj = 0; nj < 4; ++nj) {
        int col  = n0 + wn * 64 + nj * 16 + cn;
        float bv = bias[col];
        for (int mi = 0; mi < 4; ++mi) {
            int rbase = m0 + wm * 64 + mi * 16 + hi8;
            float o[8];
            for (int r = 0; r < 8; ++r) {
                o[r] = acc[mi][nj][r] + bv;
                if (relu) o[r] = fmaxf(o[r], 0.0f);
            }
            if (Cf)
                for (int r = 0; r < 8; ++r)
                    Cf[(size_t)(rbase + r) * N + col] = o[r];
            if (Cb)
                for (int r = 0; r < 8; ++r)
                    Cb[(size_t)(rbase + r) * N + col] = f2bf(o[r]);
            if (Ct) {   // transposed rows are contiguous: one 16B store
                v8u pk;
                for (int r = 0; r < 4; ++r)
                    pk[r] = (unsigned)f2bf(o[2 * r]) |
                            ((unsigned)f2bf(o[2 * r + 1]) << 16);
                uint4 q = make_uint4(pk[0], pk[1], pk[2], pk[3]);
                *reinterpret_cast<uint4*>(&Ct[(size_t)col * M + rbase]) = q;
            }
        }
    }
}

// ---------------------------------------------------------------------------
// Flash attention. q,k: bf16 token-major [B*S, D]; vt: bf16 [D, B*S]
// (vt[(h*64+d)*M + b*4096 + s]); out: bf16 token-major [B*S, D].
// One block per (batch, head, 64-query tile).
// ---------------------------------------------------------------------------
#define LDT 72
#define LDSS 65

__global__ __launch_bounds__(256)
void attn_kernel(const unsigned short* __restrict__ q,
                 const unsigned short* __restrict__ k,
                 const unsigned short* __restrict__ vt,
                 unsigned short* __restrict__ out) {
    __shared__ unsigned short q_lds[64 * LDT];
    __shared__ unsigned short k_lds[64 * LDT];
    __shared__ unsigned short vt_lds[64 * LDT];   // [d][key]
    __shared__ unsigned short p_lds[64 * LDT];
    __shared__ float s_lds[64 * LDSS];
    __shared__ float m_s[64], l_s[64], al_s[64];

    const int tid  = threadIdx.x;
    const int lane = tid & 31;
    const int wave = tid >> 5;
    const int fm   = wave & 3;
    const int fnb  = (wave >> 2) * 32;

    const int blk = blockIdx.x;
    const int qt  = blk & 63;
    const int h   = (blk >> 6) & 15;
    const int b   = blk >> 10;

    const unsigned short* qb = q  + (size_t)(b * 4096 + qt * 64) * 1024 + h * 64;
    const unsigned short* kb = k  + (size_t)(b * 4096) * 1024 + h * 64;
    const unsigned short* vb = vt + (size_t)(h * 64) * 8192 + b * 4096;
    unsigned short*       ob = out + (size_t)(b * 4096 + qt * 64) * 1024 + h * 64;

    for (int i = tid; i < 512; i += 256) {       // Q tile 64x64
        int r = i >> 3, c8 = (i & 7) << 3;
        cp16(&q_lds[r * LDT + c8], qb + (size_t)r * 1024 + c8);
    }
    if (tid < 64) { m_s[tid] = -3.0e38f; l_s[tid] = 0.0f; }
    v8f o0 = zero8(), o1 = zero8();
    async_wait_all();
    __syncthreads();

    for (int kt = 0; kt < 64; ++kt) {
        for (int i = tid; i < 512; i += 256) {   // K tile (row = key)
            int r = i >> 3, c8 = (i & 7) << 3;
            cp16(&k_lds[r * LDT + c8],
                 kb + (size_t)(kt * 64 + r) * 1024 + c8);
        }
        for (int i = tid; i < 512; i += 256) {   // Vt tile (row = d)
            int d = i >> 3, c8 = (i & 7) << 3;
            cp16(&vt_lds[d * LDT + c8],
                 vb + (size_t)d * 8192 + kt * 64 + c8);
        }
        async_wait_all();
        __syncthreads();

        // S = Q K^T (scale 1/8 applied on write-out).
        v8f s0 = zero8(), s1 = zero8();
        for (int kk = 0; kk < 64; kk += 32) {
            v16bf aq  = ld_fragA(q_lds, fm * 16, LDT, kk);
            v16bf bk0 = ld_fragB(k_lds, fnb,      LDT, kk);
            v16bf bk1 = ld_fragB(k_lds, fnb + 16, LDT, kk);
            s0 = wmma_bf16(aq, bk0, s0);
            s1 = wmma_bf16(aq, bk1, s1);
        }
        {
            int cn = lane & 15, hi8 = (lane >> 4) << 3;
            for (int r = 0; r < 8; ++r) {
                int row = fm * 16 + r + hi8;
                s_lds[row * LDSS + fnb + cn]      = 0.125f * s0[r];
                s_lds[row * LDSS + fnb + 16 + cn] = 0.125f * s1[r];
            }
        }
        __syncthreads();

        // Online softmax, one thread per query row.
        if (tid < 64) {
            float mold = m_s[tid];
            float cmax = mold;
            for (int j = 0; j < 64; ++j)
                cmax = fmaxf(cmax, s_lds[tid * LDSS + j]);
            float a = __expf(mold - cmax);
            float sum = 0.0f;
            for (int j = 0; j < 64; ++j) {
                float p = __expf(s_lds[tid * LDSS + j] - cmax);
                sum += p;
                p_lds[tid * LDT + j] = f2bf(p);
            }
            m_s[tid]  = cmax;
            l_s[tid]  = l_s[tid] * a + sum;
            al_s[tid] = a;
        }
        __syncthreads();

        // O = O*alpha + P V.
        {
            int hi8 = (lane >> 4) << 3;
            for (int r = 0; r < 8; ++r) {
                float a = al_s[fm * 16 + r + hi8];
                o0[r] *= a;
                o1[r] *= a;
            }
        }
        for (int kk = 0; kk < 64; kk += 32) {
            v16bf ap  = ld_fragA(p_lds,  fm * 16,  LDT, kk);
            v16bf bv0 = ld_fragB(vt_lds, fnb,      LDT, kk);
            v16bf bv1 = ld_fragB(vt_lds, fnb + 16, LDT, kk);
            o0 = wmma_bf16(ap, bv0, o0);
            o1 = wmma_bf16(ap, bv1, o1);
        }
        __syncthreads();
    }

    {
        int cn = lane & 15, hi8 = (lane >> 4) << 3;
        for (int r = 0; r < 8; ++r) {
            int row = fm * 16 + r + hi8;
            float inv = 1.0f / l_s[row];
            ob[(size_t)row * 1024 + fnb + cn]      = f2bf(o0[r] * inv);
            ob[(size_t)row * 1024 + fnb + 16 + cn] = f2bf(o1[r] * inv);
        }
    }
}

// ---------------------------------------------------------------------------
// outf = alpha * ((x+y) - mean) / (std_unbiased + eps) + beta; outb optional.
// ---------------------------------------------------------------------------
__global__ __launch_bounds__(256)
void add_ln_kernel(const float* __restrict__ x, const float* __restrict__ y,
                   const float* __restrict__ alpha, const float* __restrict__ beta,
                   float* __restrict__ outf, unsigned short* __restrict__ outb) {
    __shared__ float red[256];
    const int row = blockIdx.x;
    const int tid = threadIdx.x;
    const float* xr = x + (size_t)row * 1024;
    const float* yr = y + (size_t)row * 1024;

    float v[4];
    float s = 0.0f;
    for (int j = 0; j < 4; ++j) {
        v[j] = xr[tid + j * 256] + yr[tid + j * 256];
        s += v[j];
    }
    red[tid] = s;
    __syncthreads();
    for (int off = 128; off > 0; off >>= 1) {
        if (tid < off) red[tid] += red[tid + off];
        __syncthreads();
    }
    float mean = red[0] * (1.0f / 1024.0f);
    __syncthreads();

    float sq = 0.0f;
    for (int j = 0; j < 4; ++j) {
        float d = v[j] - mean;
        sq += d * d;
    }
    red[tid] = sq;
    __syncthreads();
    for (int off = 128; off > 0; off >>= 1) {
        if (tid < off) red[tid] += red[tid + off];
        __syncthreads();
    }
    float var = red[0] * (1.0f / 1023.0f);    // ddof=1
    float inv = 1.0f / (sqrtf(var) + 1e-6f);  // eps added to std

    for (int j = 0; j < 4; ++j) {
        int col = tid + j * 256;
        float o = alpha[col] * (v[j] - mean) * inv + beta[col];
        outf[(size_t)row * 1024 + col] = o;
        if (outb) outb[(size_t)row * 1024 + col] = f2bf(o);
    }
}

// ---------------------------------------------------------------------------
// Launch. Workspace: 176 MB.
// ---------------------------------------------------------------------------
extern "C" void kernel_launch(void* const* d_in, const int* in_sizes, int n_in,
                              void* d_out, int out_size, void* d_ws, size_t ws_size,
                              hipStream_t stream) {
    const float* x      = (const float*)d_in[0];
    const float* Wq     = (const float*)d_in[1];
    const float* bq     = (const float*)d_in[2];
    const float* Wk     = (const float*)d_in[3];
    const float* bk     = (const float*)d_in[4];
    const float* Wv     = (const float*)d_in[5];
    const float* bv     = (const float*)d_in[6];
    const float* Wo     = (const float*)d_in[7];
    const float* bo     = (const float*)d_in[8];
    const float* W1     = (const float*)d_in[9];
    const float* b1     = (const float*)d_in[10];
    const float* W2     = (const float*)d_in[11];
    const float* b2     = (const float*)d_in[12];
    const float* alpha1 = (const float*)d_in[13];
    const float* beta1  = (const float*)d_in[14];
    const float* alpha2 = (const float*)d_in[15];
    const float* beta2  = (const float*)d_in[16];
    float* out = (float*)d_out;

    const int NT = 8192;
    char* w = (char*)d_ws;
    const size_t MB = 1024 * 1024;
    unsigned short* x_bf      = (unsigned short*)(w + 0);        // 16 MB
    unsigned short* wqt       = (unsigned short*)(w + 16 * MB);  // 2 MB
    unsigned short* wkt       = (unsigned short*)(w + 18 * MB);
    unsigned short* wvt       = (unsigned short*)(w + 20 * MB);
    unsigned short* wot       = (unsigned short*)(w + 22 * MB);
    unsigned short* w1t       = (unsigned short*)(w + 24 * MB);  // 4 MB
    unsigned short* w2t       = (unsigned short*)(w + 28 * MB);  // 4 MB
    unsigned short* q_bf      = (unsigned short*)(w + 32 * MB);  // 16 MB
    unsigned short* k_bf      = (unsigned short*)(w + 48 * MB);  // 16 MB
    unsigned short* vt_bf     = (unsigned short*)(w + 64 * MB);  // 16 MB
    unsigned short* concat_bf = (unsigned short*)(w + 80 * MB);  // 16 MB
    float*          ln1_f     = (float*)(w + 96 * MB);           // 32 MB
    unsigned short* ln1_bf    = (unsigned short*)(w + 128 * MB); // 16 MB
    float*          attnout_f = (float*)(w + 144 * MB);          // 32 MB
    float*          ff2_f     = attnout_f;                       // reuse
    unsigned short* ff1_bf    = q_bf;                            // 32 MB reuse

    dim3 blk(256);

    // bf16 operand prep.
    cvt_bf16_kernel<<<dim3(8192), blk, 0, stream>>>(x, x_bf);
    transpose_w_kernel<<<dim3(32, 32), blk, 0, stream>>>(Wq, wqt, 1024, 1024);
    transpose_w_kernel<<<dim3(32, 32), blk, 0, stream>>>(Wk, wkt, 1024, 1024);
    transpose_w_kernel<<<dim3(32, 32), blk, 0, stream>>>(Wv, wvt, 1024, 1024);
    transpose_w_kernel<<<dim3(32, 32), blk, 0, stream>>>(Wo, wot, 1024, 1024);
    transpose_w_kernel<<<dim3(64, 32), blk, 0, stream>>>(W1, w1t, 1024, 2048);
    transpose_w_kernel<<<dim3(32, 64), blk, 0, stream>>>(W2, w2t, 2048, 1024);

    dim3 g1(1024 / BNG, NT / BMG);   // (8, 32)
    dim3 g2(2048 / BNG, NT / BMG);   // (16, 32)

    // Q, K, V projections (bf16 outputs; V directly transposed).
    gemm_bf_kernel<<<g1, blk, 0, stream>>>(x_bf, wqt, bq, nullptr, q_bf, nullptr,
                                           NT, 1024, 1024, 0);
    gemm_bf_kernel<<<g1, blk, 0, stream>>>(x_bf, wkt, bk, nullptr, k_bf, nullptr,
                                           NT, 1024, 1024, 0);
    gemm_bf_kernel<<<g1, blk, 0, stream>>>(x_bf, wvt, bv, nullptr, nullptr, vt_bf,
                                           NT, 1024, 1024, 0);

    // Flash attention: 2 batches * 16 heads * 64 q-tiles.
    attn_kernel<<<dim3(2048), blk, 0, stream>>>(q_bf, k_bf, vt_bf, concat_bf);

    // O projection (f32) + residual LN1 (f32 + bf16).
    gemm_bf_kernel<<<g1, blk, 0, stream>>>(concat_bf, wot, bo, attnout_f, nullptr,
                                           nullptr, NT, 1024, 1024, 0);
    add_ln_kernel<<<dim3(NT), blk, 0, stream>>>(x, attnout_f, alpha1, beta1,
                                                ln1_f, ln1_bf);

    // FFN.
    gemm_bf_kernel<<<g2, blk, 0, stream>>>(ln1_bf, w1t, b1, nullptr, ff1_bf,
                                           nullptr, NT, 2048, 1024, 1);
    gemm_bf_kernel<<<g1, blk, 0, stream>>>(ff1_bf, w2t, b2, ff2_f, nullptr,
                                           nullptr, NT, 1024, 2048, 0);
    add_ln_kernel<<<dim3(NT), blk, 0, stream>>>(ln1_f, ff2_f, alpha2, beta2,
                                                out, nullptr);
}
